// DAGERCModule_25692494365064
// MI455X (gfx1250) — matrix operand order repaired and checked by reading, not compiled, not run on hardware
//
#include <hip/hip_runtime.h>
#include <hip/hip_bf16.h>
#include <math.h>

typedef __attribute__((ext_vector_type(16))) _Float16 v16h;
typedef __attribute__((ext_vector_type(8)))  float    v8f;

#define HIDN  300
#define NUTT  128
#define BATCH 8
#define EMBD  100
#define NCLS  7
#define NLAY  4

// ---------------- wave reductions (wave32) ----------------
__device__ inline float wave_sum(float v){
#pragma unroll
  for (int o = 16; o > 0; o >>= 1) v += __shfl_xor(v, o, 32);
  return v;
}
__device__ inline float wave_max(float v){
#pragma unroll
  for (int o = 16; o > 0; o >>= 1) v = fmaxf(v, __shfl_xor(v, o, 32));
  return v;
}

__device__ inline v8f wmma_acc(v16h a, v16h b, v8f c){
  return __builtin_amdgcn_wmma_f32_16x16x32_f16(false, a, false, b, (short)0, c, false, false);
}

// ---- A fragment from an f32 row pointer (vector path when fully in-bounds) ----
// CDNA5 16-bit A 16x32: lane<16 -> M=lane, K {k0..k0+7, k0+16..k0+23}, k0=kb*32
//                       lane>=16 -> M=lane-16, k0=kb*32+8
__device__ inline void cvt8(v16h& r, int base, const float4 a, const float4 b){
  r[base+0]=(_Float16)a.x; r[base+1]=(_Float16)a.y; r[base+2]=(_Float16)a.z; r[base+3]=(_Float16)a.w;
  r[base+4]=(_Float16)b.x; r[base+5]=(_Float16)b.y; r[base+6]=(_Float16)b.z; r[base+7]=(_Float16)b.w;
}
__device__ inline v16h load_a_row(const float* __restrict__ p, int kb, int kvalid){
  const int lane = threadIdx.x & 31;
  const int k0 = kb*32 + ((lane & 16) ? 8 : 0);
  v16h r;
  if (k0 + 8 <= kvalid){
    cvt8(r, 0, *(const float4*)(p + k0), *(const float4*)(p + k0 + 4));
  } else {
#pragma unroll
    for (int i = 0; i < 8; ++i){ int k = k0 + i; r[i] = (k < kvalid) ? (_Float16)p[k] : (_Float16)0.f; }
  }
  const int k1 = k0 + 16;
  if (k1 + 8 <= kvalid){
    cvt8(r, 8, *(const float4*)(p + k1), *(const float4*)(p + k1 + 4));
  } else {
#pragma unroll
    for (int i = 0; i < 8; ++i){ int k = k1 + i; r[i+8] = (k < kvalid) ? (_Float16)p[k] : (_Float16)0.f; }
  }
  return r;
}
// A fragment from 8-row LDS tile (rows 8..15 are zero; K pads pre-zeroed -> unconditional vectors)
__device__ inline v16h load_a_lds8(const float* __restrict__ base, int stride, int kb){
  const int lane = threadIdx.x & 31;
  const int m = lane & 15;
  const int k0 = kb*32 + ((lane & 16) ? 8 : 0);
  v16h r;
  if (m < 8){
    const float* p = base + m*stride + k0;
    cvt8(r, 0, *(const float4*)(p),      *(const float4*)(p + 4));
    cvt8(r, 8, *(const float4*)(p + 16), *(const float4*)(p + 20));
  } else {
#pragma unroll
    for (int i = 0; i < 16; ++i) r[i] = (_Float16)0.f;
  }
  return r;
}
// packed-fragment load: one contiguous 32-byte v16h per lane
__device__ inline v16h load_packed(const _Float16* __restrict__ P, size_t tile, int KB, int kb){
  const int lane = threadIdx.x & 31;
  return *(const v16h*)(P + ((tile*(size_t)KB + kb)*32 + lane)*16);
}

// ================= weight pre-pack kernels (B layout) =================
// B 32x16: lane%16 = N col, K = kb*32 + (lane>=16?8:0) + {0..7, 16..23} per lane
// packed[(t*KB+kb)*512 + lane*16 + j] = B[k= kb*32+(lane>>4)*8+(j<8?j:j+8)][ng= t*16+(lane&15)]
__global__ __launch_bounds__(256) void k_pack_plain(const float* __restrict__ W, int Kvalid, int Nvalid,
                                                    int KB, int total, _Float16* __restrict__ out){
  int idx = blockIdx.x*256 + threadIdx.x;
  if (idx >= total) return;
  int j    = idx & 15;
  int lane = (idx >> 4) & 31;
  int kb   = (idx >> 9) % KB;
  int t    = (idx >> 9) / KB;
  int k    = kb*32 + ((lane >> 4) << 3) + (j < 8 ? j : j + 8);
  int ng   = t*16 + (lane & 15);
  float v  = (k < Kvalid && ng < Nvalid) ? W[(size_t)ng*Kvalid + k] : 0.f;
  out[idx] = (_Float16)v;
}
// stacked-K: k in [0,320) -> w0 (valid k<300); k in [320,640) -> w1. N=300, tiles=19, KB=20
__global__ __launch_bounds__(256) void k_pack_stackK(const float* __restrict__ w0, const float* __restrict__ w1,
                                                     _Float16* __restrict__ out){
  int idx = blockIdx.x*256 + threadIdx.x;
  if (idx >= 19*20*512) return;
  int j    = idx & 15;
  int lane = (idx >> 4) & 31;
  int kb   = (idx >> 9) % 20;
  int t    = (idx >> 9) / 20;
  int k    = kb*32 + ((lane >> 4) << 3) + (j < 8 ? j : j + 8);
  int ng   = t*16 + (lane & 15);
  float v = 0.f;
  if (ng < HIDN){
    if (k < 320){ if (k < HIDN) v = w0[(size_t)ng*HIDN + k]; }
    else        { int kk = k - 320; if (kk < HIDN) v = w1[(size_t)ng*HIDN + kk]; }
  }
  out[idx] = (_Float16)v;
}
// stacked-N: ng<900 -> wa; ng in [900,1800) -> wb. K=320 (valid 300), tiles=113, KB=10
__global__ __launch_bounds__(256) void k_pack_stackN(const float* __restrict__ wa, const float* __restrict__ wb,
                                                     _Float16* __restrict__ out){
  int idx = blockIdx.x*256 + threadIdx.x;
  if (idx >= 113*10*512) return;
  int j    = idx & 15;
  int lane = (idx >> 4) & 31;
  int kb   = (idx >> 9) % 10;
  int t    = (idx >> 9) / 10;
  int k    = kb*32 + ((lane >> 4) << 3) + (j < 8 ? j : j + 8);
  int ng   = t*16 + (lane & 15);
  float v = 0.f;
  if (k < HIDN && ng < 1800){
    v = (ng < 900) ? wa[(size_t)ng*HIDN + k] : wb[(size_t)(ng-900)*HIDN + k];
  }
  out[idx] = (_Float16)v;
}
// pack concat activation [H0..H4 | input] (1024 x 1600) into A-fragment layout (64 mtiles, KB=50)
__global__ __launch_bounds__(256) void k_pack_acat(const float* __restrict__ H0, const float* __restrict__ H1,
                                                   const float* __restrict__ H2, const float* __restrict__ H3,
                                                   const float* __restrict__ H4, const float* __restrict__ Xin,
                                                   _Float16* __restrict__ out){
  const float* Hs[5] = {H0, H1, H2, H3, H4};
  int idx = blockIdx.x*256 + threadIdx.x;
  if (idx >= 64*50*512) return;
  int j    = idx & 15;
  int lane = (idx >> 4) & 31;
  int kb   = (idx >> 9) % 50;
  int mt   = (idx >> 9) / 50;
  int r    = mt*16 + (lane & 15);
  int k    = kb*32 + ((lane & 16) ? 8 : 0) + (j < 8 ? j : j - 8 + 16);
  float v;
  if (k < 1500){ int s = k / HIDN; v = Hs[s][(size_t)r*HIDN + (k - s*HIDN)]; }
  else         { v = Xin[(size_t)r*EMBD + (k - 1500)]; }   // k<1600 always
  out[idx] = (_Float16)v;
}

// ================= GEMM kernels =================
// f32 A (row-major, ld=Kvalid) x packed-B -> Y(1024,NOUT), optional relu
template <int KB, int NT, int NOUT, bool RELU>
__global__ __launch_bounds__(256) void k_gemm_pb(const float* __restrict__ X, int kvalid,
                                                 const _Float16* __restrict__ Bp,
                                                 const float* __restrict__ bias,
                                                 float* __restrict__ Y){
  const int mt   = blockIdx.x;
  const int wave = threadIdx.x >> 5;
  const int lane = threadIdx.x & 31;
  const float* row = X + (size_t)(mt*16 + (lane & 15))*kvalid;
  for (int nt = wave; nt < NT; nt += 8){
    v8f acc = {0,0,0,0,0,0,0,0};
    for (int kb = 0; kb < KB; ++kb){
      v16h af = load_a_row(row, kb, kvalid);
      v16h bf = load_packed(Bp, nt, KB, kb);
      acc = wmma_acc(af, bf, acc);
    }
    int col = nt*16 + (lane & 15);
    if (col < NOUT){
      int r0 = mt*16 + ((lane & 16) ? 8 : 0);
#pragma unroll
      for (int e = 0; e < 8; ++e){
        float v = acc[e] + bias[col];
        if (RELU) v = v > 0.f ? v : 0.f;
        Y[(size_t)(r0 + e)*NOUT + col] = v;
      }
    }
  }
}
// packed-A x packed-B (mlp1): KB=50, NT=19, relu
__global__ __launch_bounds__(256) void k_gemm_pp(const _Float16* __restrict__ Ap,
                                                 const _Float16* __restrict__ Bp,
                                                 const float* __restrict__ bias,
                                                 float* __restrict__ Y){
  const int mt   = blockIdx.x;
  const int wave = threadIdx.x >> 5;
  const int lane = threadIdx.x & 31;
  for (int nt = wave; nt < 19; nt += 8){
    v8f acc = {0,0,0,0,0,0,0,0};
    for (int kb = 0; kb < 50; ++kb){
      v16h af = load_packed(Ap, mt, 50, kb);
      v16h bf = load_packed(Bp, nt, 50, kb);
      acc = wmma_acc(af, bf, acc);
    }
    int col = nt*16 + (lane & 15);
    if (col < HIDN){
      int r0 = mt*16 + ((lane & 16) ? 8 : 0);
#pragma unroll
      for (int e = 0; e < 8; ++e){
        float v = acc[e] + bias[col];
        Y[(size_t)(r0 + e)*HIDN + col] = v > 0.f ? v : 0.f;
      }
    }
  }
}

// ================= DAG layer scan: persistent single-workgroup kernel =================
__global__ __launch_bounds__(1024) void k_scan(
    const float* __restrict__ Hin, float* __restrict__ Hout,
    const int* __restrict__ adj, const int* __restrict__ smask,
    const float* __restrict__ gw, const float* __restrict__ gbp,
    const _Float16* __restrict__ pMAT,   // packed [w0;w1]   (19 tiles, KB=20)
    const _Float16* __restrict__ pX,     // packed [cwih|pwhh] (113 tiles, KB=10)
    const _Float16* __restrict__ pM,     // packed [cwhh|pwih] (113 tiles, KB=10)
    const float* __restrict__ cbih, const float* __restrict__ cbhh,
    const float* __restrict__ pbih, const float* __restrict__ pbhh,
    float* __restrict__ gX, float* __restrict__ gM)
{
  __shared__ float xs[BATCH][320];
  __shared__ float Ms[BATCH][320];
  __shared__ float a2[BATCH][640];       // [m0 | m1], each padded to 320
  __shared__ float attn[BATCH][NUTT];
  __shared__ float hdot[BATCH][NUTT];
  __shared__ float hrow[BATCH][HIDN];
  __shared__ float xdot[BATCH];

  const int tid  = threadIdx.x;
  const int wave = tid >> 5;
  const int lane = tid & 31;
  const float gb = gbp[0];

  // zero K-padding columns once
  for (int t = tid; t < BATCH*20; t += 1024){
    int b = t / 20, c = 300 + (t % 20);
    xs[b][c] = 0.f; Ms[b][c] = 0.f; a2[b][c] = 0.f; a2[b][c + 320] = 0.f;
  }
  __syncthreads();

  for (int i = 0; i < NUTT; ++i){
    // ---- stage 1: load x_i into LDS (float4) ----
    for (int t = tid; t < BATCH*75; t += 1024){
      int b = t / 75, d0 = (t - b*75)*4;
      *(float4*)&xs[b][d0] = *(const float4*)(Hin + (size_t)(b*NUTT + i)*HIDN + d0);
    }
    __syncthreads();
    if (wave < BATCH){
      int b = wave; float p = 0.f;
      for (int d = lane; d < HIDN; d += 32) p += xs[b][d] * gw[d];
      p = wave_sum(p);
      if (lane == 0) xdot[b] = p;
    }
    __syncthreads();

    if (i > 0){
      // ---- stage 2: masked softmax over j < i (one wave per batch row) ----
      if (wave < BATCH){
        int b = wave;
        float mx = -3.0e38f;
        for (int j = lane; j < i; j += 32){
          int v = adj[(size_t)(b*NUTT + i)*NUTT + j];
          float a = v ? (xdot[b] + hdot[b][j] + gb) : -1.0e30f;
          attn[b][j] = a;
          mx = fmaxf(mx, a);
        }
        mx = wave_max(mx);
        float s = 0.f;
        for (int j = lane; j < i; j += 32){
          float e = expf(attn[b][j] - mx);
          attn[b][j] = e;
          s += e;
        }
        s = wave_sum(s);
        float inv = 1.f / s;
        for (int j = lane; j < i; j += 32) attn[b][j] *= inv;
      }
      __syncthreads();
      // ---- stage 3: m0/m1 weighted sums (float4 per thread) ----
      for (int t = tid; t < BATCH*75; t += 1024){
        int b = t / 75, d0 = (t - b*75)*4;
        float4 m0 = {0,0,0,0}, m1 = {0,0,0,0};
        const float* hp = Hout + (size_t)b*NUTT*HIDN + d0;
        const int*   sp = smask + (size_t)(b*NUTT + i)*NUTT;
        for (int j = 0; j < i; ++j){
          float a  = attn[b][j];
          float sm = (float)sp[j];
          float as = a*sm, ad = a - as;
          float4 h = *(const float4*)(hp + (size_t)j*HIDN);
          m0.x += as*h.x; m0.y += as*h.y; m0.z += as*h.z; m0.w += as*h.w;
          m1.x += ad*h.x; m1.y += ad*h.y; m1.z += ad*h.z; m1.w += ad*h.w;
        }
        *(float4*)&a2[b][d0]       = m0;
        *(float4*)&a2[b][320 + d0] = m1;
      }
      __syncthreads();
      // ---- stage 4: M = m0 @ w0^T + m1 @ w1^T (stacked K=640) ----
      for (int nt = wave; nt < 19; nt += 32){
        v8f acc = {0,0,0,0,0,0,0,0};
        for (int kb = 0; kb < 20; ++kb){
          v16h af = load_a_lds8(&a2[0][0], 640, kb);
          v16h bf = load_packed(pMAT, nt, 20, kb);
          acc = wmma_acc(af, bf, acc);
        }
        if (lane < 16){
          int ng = nt*16 + lane;
          if (ng < HIDN){
#pragma unroll
            for (int e = 0; e < 8; ++e) Ms[e][ng] = acc[e];
          }
        }
      }
      __syncthreads();
    } else {
      for (int t = tid; t < BATCH*HIDN; t += 1024){
        int b = t / HIDN, d = t - b*HIDN;
        Ms[b][d] = 0.f;
      }
      __syncthreads();
    }

    // ---- stage 5: gX = x @ [cwih|pwhh]^T ; gM = M @ [cwhh|pwih]^T ----
    for (int t = wave; t < 226; t += 32){
      const int isX = (t < 113) ? 1 : 0;
      const int nt  = isX ? t : (t - 113);
      const float*    Abase = isX ? &xs[0][0] : &Ms[0][0];
      const _Float16* Bp    = isX ? pX : pM;
      v8f acc = {0,0,0,0,0,0,0,0};
      for (int kb = 0; kb < 10; ++kb){
        v16h af = load_a_lds8(Abase, 320, kb);
        v16h bf = load_packed(Bp, nt, 10, kb);
        acc = wmma_acc(af, bf, acc);
      }
      if (lane < 16){
        int ng = nt*16 + lane;
        if (ng < 1800){
          float* out = isX ? gX : gM;
#pragma unroll
          for (int e = 0; e < 8; ++e) out[ng*8 + e] = acc[e];
        }
      }
    }
    __threadfence();
    __syncthreads();

    // ---- stage 6: GRU gate combine; H1[:,i,:] = C + P ----
    for (int t = tid; t < BATCH*HIDN; t += 1024){
      int b = t / HIDN, d = t - b*HIDN;
      float x  = xs[b][d], Mv = Ms[b][d];
      float ir = gX[(d       )*8 + b] + cbih[d];
      float iz = gX[(d +  300)*8 + b] + cbih[d + 300];
      float in_= gX[(d +  600)*8 + b] + cbih[d + 600];
      float hr = gM[(d       )*8 + b] + cbhh[d];
      float hz = gM[(d +  300)*8 + b] + cbhh[d + 300];
      float hn = gM[(d +  600)*8 + b] + cbhh[d + 600];
      float r  = 1.f / (1.f + expf(-(ir + hr)));
      float z  = 1.f / (1.f + expf(-(iz + hz)));
      float nn = tanhf(in_ + r*hn);
      float C  = (1.f - z)*nn + z*Mv;
      float pir = gM[(d +  900)*8 + b] + pbih[d];
      float piz = gM[(d + 1200)*8 + b] + pbih[d + 300];
      float pin = gM[(d + 1500)*8 + b] + pbih[d + 600];
      float phr = gX[(d +  900)*8 + b] + pbhh[d];
      float phz = gX[(d + 1200)*8 + b] + pbhh[d + 300];
      float phn = gX[(d + 1500)*8 + b] + pbhh[d + 600];
      float pr  = 1.f / (1.f + expf(-(pir + phr)));
      float pz  = 1.f / (1.f + expf(-(piz + phz)));
      float pn  = tanhf(pin + pr*phn);
      float P   = (1.f - pz)*pn + pz*x;
      float h = C + P;
      Hout[(size_t)(b*NUTT + i)*HIDN + d] = h;
      hrow[b][d] = h;
    }
    __threadfence();
    __syncthreads();
    // ---- stage 7: cache hdot[b][i] ----
    if (wave < BATCH){
      int b = wave; float p = 0.f;
      for (int d = lane; d < HIDN; d += 32) p += hrow[b][d] * gw[HIDN + d];
      p = wave_sum(p);
      if (lane == 0) hdot[b][i] = p;
    }
    __syncthreads();
  }
}

// ================= host launch =================
extern "C" void kernel_launch(void* const* d_in, const int* in_sizes, int n_in,
                              void* d_out, int out_size, void* d_ws, size_t ws_size,
                              hipStream_t stream) {
  (void)in_sizes; (void)n_in; (void)out_size; (void)ws_size;
  const float* input   = (const float*)d_in[0];
  const int*   adj     = (const int*)  d_in[1];
  const int*   smask   = (const int*)  d_in[2];
  const float* fc1_w   = (const float*)d_in[4];
  const float* fc1_b   = (const float*)d_in[5];
  const float* gat_w   = (const float*)d_in[6];
  const float* gat_b   = (const float*)d_in[7];
  const float* wr0     = (const float*)d_in[8];
  const float* wr1     = (const float*)d_in[9];
  const float* gc_wih  = (const float*)d_in[10];
  const float* gc_whh  = (const float*)d_in[11];
  const float* gc_bih  = (const float*)d_in[12];
  const float* gc_bhh  = (const float*)d_in[13];
  const float* gp_wih  = (const float*)d_in[14];
  const float* gp_whh  = (const float*)d_in[15];
  const float* gp_bih  = (const float*)d_in[16];
  const float* gp_bhh  = (const float*)d_in[17];
  const float* mlp1_w  = (const float*)d_in[18];
  const float* mlp1_b  = (const float*)d_in[19];
  const float* mlp2_w  = (const float*)d_in[20];
  const float* mlp2_b  = (const float*)d_in[21];
  const float* mlp3_w  = (const float*)d_in[22];
  const float* mlp3_b  = (const float*)d_in[23];

  float* ws = (float*)d_ws;
  const size_t HSZ = (size_t)BATCH * NUTT * HIDN;     // 307200
  float* Hs[NLAY + 1];
  for (int l = 0; l <= NLAY; ++l) Hs[l] = ws + (size_t)l * HSZ;
  float* h1 = ws + 5*HSZ;
  float* h2 = h1 + HSZ;
  float* gX = h2 + HSZ;                 // 1808*8 floats
  float* gM = gX + 1808*8;
  float* fend = gM + 1808*8;            // end of f32 region (2,179,328 floats)

  // f16 packed region
  _Float16* hp = (_Float16*)fend;
  const size_t SZ_MAT = 19*20*512;      // 194560
  const size_t SZ_SN  = 113*10*512;     // 578560
  const size_t SZ_LAY = SZ_MAT + 2*SZ_SN;
  _Float16* pMAT[NLAY], *pXL[NLAY], *pML[NLAY];
  for (int l = 0; l < NLAY; ++l){
    pMAT[l] = hp + (size_t)l*SZ_LAY;
    pXL[l]  = pMAT[l] + SZ_MAT;
    pML[l]  = pXL[l]  + SZ_SN;
  }
  _Float16* pfc1  = hp + (size_t)NLAY*SZ_LAY;
  const size_t SZ_FC1 = 19*4*512, SZ_M1 = 19*50*512, SZ_M2 = 19*10*512, SZ_M3 = 1*10*512;
  _Float16* pmlp1 = pfc1  + SZ_FC1;
  _Float16* pmlp2 = pmlp1 + SZ_M1;
  _Float16* pmlp3 = pmlp2 + SZ_M2;
  _Float16* pacat = pmlp3 + SZ_M3;      // 64*50*512 halves

  // ---- pre-pack weights into WMMA fragment layouts (runs once per call) ----
  auto grid = [](size_t n){ return (unsigned)((n + 255) / 256); };
  k_pack_plain<<<grid(SZ_FC1), 256, 0, stream>>>(fc1_w, EMBD, HIDN, 4, (int)SZ_FC1, pfc1);
  k_pack_plain<<<grid(SZ_M1),  256, 0, stream>>>(mlp1_w, 1600, HIDN, 50, (int)SZ_M1, pmlp1);
  k_pack_plain<<<grid(SZ_M2),  256, 0, stream>>>(mlp2_w, HIDN, HIDN, 10, (int)SZ_M2, pmlp2);
  k_pack_plain<<<grid(SZ_M3),  256, 0, stream>>>(mlp3_w, HIDN, NCLS, 10, (int)SZ_M3, pmlp3);
  for (int l = 0; l < NLAY; ++l){
    k_pack_stackK<<<grid(SZ_MAT), 256, 0, stream>>>(wr0 + (size_t)l*HIDN*HIDN,
                                                    wr1 + (size_t)l*HIDN*HIDN, pMAT[l]);
    k_pack_stackN<<<grid(SZ_SN), 256, 0, stream>>>(gc_wih + (size_t)l*900*HIDN,
                                                   gp_whh + (size_t)l*900*HIDN, pXL[l]);
    k_pack_stackN<<<grid(SZ_SN), 256, 0, stream>>>(gc_whh + (size_t)l*900*HIDN,
                                                   gp_wih + (size_t)l*900*HIDN, pML[l]);
  }

  // H0 = relu(input @ fc1^T + b)
  k_gemm_pb<4, 19, HIDN, true><<<64, 256, 0, stream>>>(input, EMBD, pfc1, fc1_b, Hs[0]);

  // 4 sequential DAG layers
  for (int l = 0; l < NLAY; ++l){
    k_scan<<<1, 1024, 0, stream>>>(Hs[l], Hs[l+1], adj, smask,
                                   gat_w + l*600, gat_b + l,
                                   pMAT[l], pXL[l], pML[l],
                                   gc_bih + (size_t)l*900, gc_bhh + (size_t)l*900,
                                   gp_bih + (size_t)l*900, gp_bhh + (size_t)l*900,
                                   gX, gM);
  }

  // pack concat [H0..H4 | input] into A fragments, then MLP head
  k_pack_acat<<<grid(64*50*512), 256, 0, stream>>>(Hs[0], Hs[1], Hs[2], Hs[3], Hs[4], input, pacat);
  k_gemm_pp<<<64, 256, 0, stream>>>(pacat, pmlp1, mlp1_b, h1);
  k_gemm_pb<10, 19, HIDN, true><<<64, 256, 0, stream>>>(h1, HIDN, pmlp2, mlp2_b, h2);
  k_gemm_pb<10, 1, NCLS, false><<<64, 256, 0, stream>>>(h2, HIDN, pmlp3, mlp3_b, (float*)d_out);
}
